// Transformer_10230612099870
// MI455X (gfx1250) — matrix operand order repaired and checked by reading, not compiled, not run on hardware
//
#include <hip/hip_runtime.h>

// ---------------------------------------------------------------------------
// MI455X (gfx1250) transformer forward: bf16 WMMA everywhere, fp32 residual.
//   B=2, S=1024, D=1024, H=16, dh=64, L=4, C=1000.  ~116 GFLOP, compute-bound.
// ---------------------------------------------------------------------------

#define Bsz 2
#define Seq 1024
#define Dm 1024
#define NH 16
#define DH 64
#define NLAYER 4
#define Ncls 1000

typedef __attribute__((ext_vector_type(16))) __bf16 bf16x16;
typedef __attribute__((ext_vector_type(8)))  float  f32x8;

union FragU { bf16x16 v; unsigned int u[8]; unsigned short s[16]; };

__device__ __forceinline__ unsigned short f2bf(float f) {
  unsigned int x = __builtin_bit_cast(unsigned int, f);
  x += 0x7FFFu + ((x >> 16) & 1u);          // round-to-nearest-even
  return (unsigned short)(x >> 16);
}

// A-operand fragment (16x32 bf16): lane-half h reads contiguous K in
// [h*16, h*16+16) from a K-contiguous row.
__device__ __forceinline__ bf16x16 load_fragA(const unsigned short* __restrict__ p, int half) {
  FragU f;
  const uint4 c1 = *reinterpret_cast<const uint4*>(p + half * 16);
  const uint4 c2 = *reinterpret_cast<const uint4*>(p + half * 16 + 8);
  f.u[0] = c1.x; f.u[1] = c1.y; f.u[2] = c1.z; f.u[3] = c1.w;
  f.u[4] = c2.x; f.u[5] = c2.y; f.u[6] = c2.z; f.u[7] = c2.w;
  return f.v;
}

// B-operand fragment (32x16 bf16): lane-half h reads contiguous K in
// [h*16, h*16+16) from a K-contiguous row (column of B = row of W^T).
__device__ __forceinline__ bf16x16 load_fragB(const unsigned short* __restrict__ p, int half) {
  FragU f;
  const uint4 c1 = *reinterpret_cast<const uint4*>(p + half * 16);
  const uint4 c2 = *reinterpret_cast<const uint4*>(p + half * 16 + 8);
  f.u[0] = c1.x; f.u[1] = c1.y; f.u[2] = c1.z; f.u[3] = c1.w;
  f.u[4] = c2.x; f.u[5] = c2.y; f.u[6] = c2.z; f.u[7] = c2.w;
  return f.v;
}

#define WMMA_BF16(A, B, C) \
  __builtin_amdgcn_wmma_f32_16x16x32_bf16(false, (A), false, (B), (short)0, (C), false, false)

// ---------------------------------------------------------------------------
// fp32 -> bf16 conversion (weights, one-time per launch)
// ---------------------------------------------------------------------------
__global__ __launch_bounds__(256) void cvt_f32_bf16(const float* __restrict__ s,
                                                    unsigned short* __restrict__ d, int n) {
  int i = (blockIdx.x * 256 + threadIdx.x) * 4;
  if (i + 3 < n) {
    float4 v = *reinterpret_cast<const float4*>(s + i);
    uint2 o;
    o.x = (unsigned)f2bf(v.x) | ((unsigned)f2bf(v.y) << 16);
    o.y = (unsigned)f2bf(v.z) | ((unsigned)f2bf(v.w) << 16);
    *reinterpret_cast<uint2*>(d + i) = o;
  } else {
    for (int j = 0; j < 4; ++j)
      if (i + j < n) d[i + j] = f2bf(s[i + j]);
  }
}

// ---------------------------------------------------------------------------
// LayerNorm (torch style: ddof=1 std, eps added to std), fp32 in -> bf16 out
// one block (256 threads) per row of D=1024
// ---------------------------------------------------------------------------
__global__ __launch_bounds__(256) void layernorm_bf16(const float* __restrict__ X,
                                                      const float* __restrict__ g,
                                                      const float* __restrict__ b,
                                                      unsigned short* __restrict__ Y) {
  const int row = blockIdx.x;
  const int t = threadIdx.x;
  const float4 x = reinterpret_cast<const float4*>(X + (size_t)row * Dm)[t];
  float s1 = x.x + x.y + x.z + x.w;
  float s2 = x.x * x.x + x.y * x.y + x.z * x.z + x.w * x.w;
  for (int off = 16; off >= 1; off >>= 1) {
    s1 += __shfl_xor(s1, off);
    s2 += __shfl_xor(s2, off);
  }
  __shared__ float ps1[8], ps2[8];
  if ((t & 31) == 0) { ps1[t >> 5] = s1; ps2[t >> 5] = s2; }
  __syncthreads();
  s1 = 0.f; s2 = 0.f;
#pragma unroll
  for (int i = 0; i < 8; ++i) { s1 += ps1[i]; s2 += ps2[i]; }
  const float mean = s1 * (1.0f / Dm);
  float var = (s2 - (float)Dm * mean * mean) * (1.0f / (Dm - 1));
  var = fmaxf(var, 0.0f);
  const float inv = 1.0f / (sqrtf(var) + 1e-6f);
  const float4 gv = reinterpret_cast<const float4*>(g)[t];
  const float4 bv = reinterpret_cast<const float4*>(b)[t];
  uint2 o;
  o.x = (unsigned)f2bf(gv.x * (x.x - mean) * inv + bv.x) |
        ((unsigned)f2bf(gv.y * (x.y - mean) * inv + bv.y) << 16);
  o.y = (unsigned)f2bf(gv.z * (x.z - mean) * inv + bv.z) |
        ((unsigned)f2bf(gv.w * (x.w - mean) * inv + bv.w) << 16);
  reinterpret_cast<uint2*>(Y + (size_t)row * Dm)[t] = o;
}

// ---------------------------------------------------------------------------
// GEMM (NT): out[m,n] = sum_k A[m,k] * W[n,k] + bias[n] (+resid) (+relu)
// A,W bf16; output fp32 or bf16. Wave tile 32(M) x 64(N), block = 8 waves
// covering 256(M) x 64(N). K stepped by 32 -> 8 v_wmma per iteration.
// ---------------------------------------------------------------------------
__global__ __launch_bounds__(256) void gemm_bf16_nt(
    const unsigned short* __restrict__ A,   // [M,K] bf16
    const unsigned short* __restrict__ W,   // [N,K] bf16
    const float* __restrict__ bias,         // [N]
    const float* __restrict__ resid,        // [M,N] fp32 or nullptr
    float* __restrict__ outF,               // [M,N] fp32 or nullptr
    unsigned short* __restrict__ outB,      // [M,N] bf16 or nullptr
    int M, int N, int K, int relu) {
  const int lane = threadIdx.x;
  const int w = threadIdx.y;
  const int half = lane >> 4;
  const int q = lane & 15;
  const int m0 = blockIdx.y * 256 + w * 32;
  const int n0 = blockIdx.x * 64;

  const f32x8 zero = {};
  f32x8 acc[2][4];
#pragma unroll
  for (int i = 0; i < 2; ++i)
#pragma unroll
    for (int t = 0; t < 4; ++t) acc[i][t] = zero;

  const unsigned short* a0p = A + (size_t)(m0 + q) * K;
  const unsigned short* a1p = A + (size_t)(m0 + 16 + q) * K;
  const unsigned short* wp[4];
#pragma unroll
  for (int t = 0; t < 4; ++t) {
    int nn = n0 + 16 * t + q;
    if (nn > N - 1) nn = N - 1;   // clamp for N=1000 edge tiles
    wp[t] = W + (size_t)nn * K;
  }

  for (int kk = 0; kk < K; kk += 32) {
    const bf16x16 af0 = load_fragA(a0p + kk, half);
    const bf16x16 af1 = load_fragA(a1p + kk, half);
#pragma unroll
    for (int t = 0; t < 4; ++t) {
      const bf16x16 bf = load_fragB(wp[t] + kk, half);
      acc[0][t] = WMMA_BF16(af0, bf, acc[0][t]);
      acc[1][t] = WMMA_BF16(af1, bf, acc[1][t]);
    }
  }

#pragma unroll
  for (int mi = 0; mi < 2; ++mi) {
#pragma unroll
    for (int t = 0; t < 4; ++t) {
      const int n = n0 + 16 * t + q;
      if (n < N) {
        const float bn = bias[n];
#pragma unroll
        for (int r = 0; r < 8; ++r) {
          const int row = m0 + mi * 16 + r + 8 * half;   // C/D VGPR layout
          float v = acc[mi][t][r] + bn;
          if (resid) v += resid[(size_t)row * N + n];
          if (relu) v = fmaxf(v, 0.0f);
          if (outB) outB[(size_t)row * N + n] = f2bf(v);
          else      outF[(size_t)row * N + n] = v;
        }
      }
    }
  }
}

// ---------------------------------------------------------------------------
// Flash attention (causal). One wave per 16-query tile of one (b,h).
// Key blocks of 32: S = Q K^T via 4 WMMA, online softmax, P bounced through
// LDS for C->A re-layout, O += P V via 4 WMMA. All of Q/K/V/O bf16.
// ---------------------------------------------------------------------------
__global__ __launch_bounds__(256) void flash_attn(const unsigned short* __restrict__ Q,
                                                  const unsigned short* __restrict__ Kx,
                                                  const unsigned short* __restrict__ V,
                                                  unsigned short* __restrict__ O) {
  __shared__ unsigned short ldsP[8][16][40];   // per-wave 16x32 P tile, padded
  const int lane = threadIdx.x;
  const int w = threadIdx.y;
  const int half = lane >> 4;
  const int qn = lane & 15;
  const int qb = blockIdx.x * 8 + w;           // query block 0..63
  const int b = blockIdx.y >> 4;
  const int h = blockIdx.y & 15;
  const int q0 = qb * 16;

  const unsigned short* qrow = Q + ((size_t)(b * Seq + q0 + qn)) * Dm + h * DH;
  const bf16x16 aq0 = load_fragA(qrow, half);       // d = 0..31
  const bf16x16 aq1 = load_fragA(qrow + 32, half);  // d = 32..63

  const f32x8 zero = {};
  f32x8 accO[4] = {zero, zero, zero, zero};
  float rm[8], rl[8];
#pragma unroll
  for (int r = 0; r < 8; ++r) { rm[r] = -1e30f; rl[r] = 0.0f; }

  const int nkb = ((q0 + 15) >> 5) + 1;   // key blocks touching causal region
  for (int kb = 0; kb < nkb; ++kb) {
    const int k0 = kb * 32;
    f32x8 sc[2] = {zero, zero};
#pragma unroll
    for (int kt = 0; kt < 2; ++kt) {
      const unsigned short* krow = Kx + ((size_t)(b * Seq + k0 + kt * 16 + qn)) * Dm + h * DH;
      const bf16x16 bk0 = load_fragB(krow, half);
      const bf16x16 bk1 = load_fragB(krow + 32, half);
      sc[kt] = WMMA_BF16(aq0, bk0, sc[kt]);
      sc[kt] = WMMA_BF16(aq1, bk1, sc[kt]);
    }

    float p0[8], p1[8], mx[8];
#pragma unroll
    for (int r = 0; r < 8; ++r) {
      const int row = q0 + r + 8 * half;          // query index for this slot
      float v0 = sc[0][r] * 0.125f;               // 1/sqrt(64)
      float v1 = sc[1][r] * 0.125f;
      if (k0 + qn > row)      v0 = -1e30f;        // causal mask
      if (k0 + 16 + qn > row) v1 = -1e30f;
      p0[r] = v0; p1[r] = v1;
      float m = fmaxf(v0, v1);                    // row-max across 16 lanes
      m = fmaxf(m, __shfl_xor(m, 1));
      m = fmaxf(m, __shfl_xor(m, 2));
      m = fmaxf(m, __shfl_xor(m, 4));
      m = fmaxf(m, __shfl_xor(m, 8));
      mx[r] = m;
    }
#pragma unroll
    for (int r = 0; r < 8; ++r) {
      const float mn = fmaxf(rm[r], mx[r]);
      const float alpha = __expf(rm[r] - mn);
      rm[r] = mn;
      p0[r] = __expf(p0[r] - mn);
      p1[r] = __expf(p1[r] - mn);
      float ps = p0[r] + p1[r];
      ps += __shfl_xor(ps, 1);
      ps += __shfl_xor(ps, 2);
      ps += __shfl_xor(ps, 4);
      ps += __shfl_xor(ps, 8);
      rl[r] = rl[r] * alpha + ps;
#pragma unroll
      for (int t = 0; t < 4; ++t) accO[t][r] *= alpha;
      ldsP[w][r + 8 * half][qn]      = f2bf(p0[r]);
      ldsP[w][r + 8 * half][16 + qn] = f2bf(p1[r]);
    }
    // same-wave DS ops are in-order; wait for writes before re-layout reads
    asm volatile("s_wait_dscnt 0" ::: "memory");
    FragU pf;
    {
      const unsigned short* pr = &ldsP[w][qn][0];
      const uint4 c1 = *reinterpret_cast<const uint4*>(pr + half * 16);
      const uint4 c2 = *reinterpret_cast<const uint4*>(pr + half * 16 + 8);
      pf.u[0] = c1.x; pf.u[1] = c1.y; pf.u[2] = c1.z; pf.u[3] = c1.w;
      pf.u[4] = c2.x; pf.u[5] = c2.y; pf.u[6] = c2.z; pf.u[7] = c2.w;
    }
#pragma unroll
    for (int t = 0; t < 4; ++t) {
      FragU vf;
#pragma unroll
      for (int e = 0; e < 16; ++e) {
        const int k = k0 + half * 16 + e;   // B-layout: contiguous K per half
        vf.s[e] = V[((size_t)(b * Seq + k)) * Dm + h * DH + t * 16 + qn];
      }
      accO[t] = WMMA_BF16(pf.v, vf.v, accO[t]);
    }
  }

#pragma unroll
  for (int t = 0; t < 4; ++t) {
#pragma unroll
    for (int r = 0; r < 8; ++r) {
      const int row = q0 + r + 8 * half;
      O[((size_t)(b * Seq + row)) * Dm + h * DH + t * 16 + qn] = f2bf(accO[t][r] / rl[r]);
    }
  }
}

// ---------------------------------------------------------------------------
// Orchestration
// ---------------------------------------------------------------------------
extern "C" void kernel_launch(void* const* d_in, const int* in_sizes, int n_in,
                              void* d_out, int out_size, void* d_ws, size_t ws_size,
                              hipStream_t stream) {
  (void)in_sizes; (void)n_in; (void)out_size; (void)ws_size;
  const float* x     = (const float*)d_in[0];
  const float* ln_a  = (const float*)d_in[1];
  const float* ln_b  = (const float*)d_in[2];
  const float* Wq    = (const float*)d_in[3];
  const float* bq    = (const float*)d_in[4];
  const float* Wk    = (const float*)d_in[5];
  const float* bk    = (const float*)d_in[6];
  const float* Wv    = (const float*)d_in[7];
  const float* bv    = (const float*)d_in[8];
  const float* Wo    = (const float*)d_in[9];
  const float* bo    = (const float*)d_in[10];
  const float* lnf_a = (const float*)d_in[11];
  const float* lnf_b = (const float*)d_in[12];
  const float* W1    = (const float*)d_in[13];
  const float* b1    = (const float*)d_in[14];
  const float* W2    = (const float*)d_in[15];
  const float* b2    = (const float*)d_in[16];
  const float* W3    = (const float*)d_in[17];
  const float* b3    = (const float*)d_in[18];

  const int M = Bsz * Seq;                       // 2048
  const size_t MD  = (size_t)M * Dm;             // 2M elements
  const size_t DD  = (size_t)Dm * Dm;            // 1M
  const size_t WDD = (size_t)NLAYER * DD;        // 4M
  const size_t CD  = (size_t)Ncls * Dm;

  // workspace carve (fp32 residual stream + bf16 activations + bf16 weights)
  char* ws = (char*)d_ws;
  float* X = (float*)ws;                         ws += MD * sizeof(float);
  unsigned short* Hb  = (unsigned short*)ws;     ws += MD * 2;
  unsigned short* Qb  = (unsigned short*)ws;     ws += MD * 2;
  unsigned short* Kb2 = (unsigned short*)ws;     ws += MD * 2;
  unsigned short* Vb2 = (unsigned short*)ws;     ws += MD * 2;
  unsigned short* Ob  = (unsigned short*)ws;     ws += MD * 2;
  unsigned short* WqB = (unsigned short*)ws;     ws += WDD * 2;
  unsigned short* WkB = (unsigned short*)ws;     ws += WDD * 2;
  unsigned short* WvB = (unsigned short*)ws;     ws += WDD * 2;
  unsigned short* WoB = (unsigned short*)ws;     ws += WDD * 2;
  unsigned short* W1B = (unsigned short*)ws;     ws += DD * 2;
  unsigned short* W2B = (unsigned short*)ws;     ws += DD * 2;
  unsigned short* W3B = (unsigned short*)ws;     ws += CD * 2;

  auto cvt = [&](const float* s, unsigned short* d, size_t n) {
    cvt_f32_bf16<<<dim3((unsigned)((n + 1023) / 1024)), dim3(256), 0, stream>>>(s, d, (int)n);
  };
  cvt(Wq, WqB, WDD);  cvt(Wk, WkB, WDD);  cvt(Wv, WvB, WDD);  cvt(Wo, WoB, WDD);
  cvt(W1, W1B, DD);   cvt(W2, W2B, DD);   cvt(W3, W3B, CD);
  hipMemcpyAsync(X, x, MD * sizeof(float), hipMemcpyDeviceToDevice, stream);

  const dim3 gblk(32, 8);
  const dim3 ggrid(Dm / 64, M / 256);            // (16, 8)

  for (int l = 0; l < NLAYER; ++l) {
    layernorm_bf16<<<M, 256, 0, stream>>>(X, ln_a + (size_t)l * Dm, ln_b + (size_t)l * Dm, Hb);
    gemm_bf16_nt<<<ggrid, gblk, 0, stream>>>(Hb, WqB + (size_t)l * DD, bq + (size_t)l * Dm,
                                             nullptr, nullptr, Qb, M, Dm, Dm, 0);
    gemm_bf16_nt<<<ggrid, gblk, 0, stream>>>(Hb, WkB + (size_t)l * DD, bk + (size_t)l * Dm,
                                             nullptr, nullptr, Kb2, M, Dm, Dm, 0);
    gemm_bf16_nt<<<ggrid, gblk, 0, stream>>>(Hb, WvB + (size_t)l * DD, bv + (size_t)l * Dm,
                                             nullptr, nullptr, Vb2, M, Dm, Dm, 0);
    flash_attn<<<dim3(Seq / 16 / 8, Bsz * NH), gblk, 0, stream>>>(Qb, Kb2, Vb2, Ob);
    gemm_bf16_nt<<<ggrid, gblk, 0, stream>>>(Ob, WoB + (size_t)l * DD, bo + (size_t)l * Dm,
                                             X, X, nullptr, M, Dm, Dm, 0);   // residual add
  }

  layernorm_bf16<<<M, 256, 0, stream>>>(X, lnf_a, lnf_b, Hb);
  gemm_bf16_nt<<<ggrid, gblk, 0, stream>>>(Hb, W1B, b1, nullptr, nullptr, Qb, M, Dm, Dm, 1);
  gemm_bf16_nt<<<ggrid, gblk, 0, stream>>>(Qb, W2B, b2, nullptr, nullptr, Kb2, M, Dm, Dm, 1);
  gemm_bf16_nt<<<dim3((Ncls + 63) / 64, M / 256), gblk, 0, stream>>>(
      Kb2, W3B, b3, nullptr, (float*)d_out, nullptr, M, Ncls, Dm, 0);
}
